// SimpleBaselineRemoval_41893111005596
// MI455X (gfx1250) — matrix operand order repaired and checked by reading, not compiled, not run on hardware
//
#include <hip/hip_runtime.h>
#include <math.h>

// ---------------------------------------------------------------------------
// SimpleBaselineRemoval for MI455X (gfx1250, wave32, WMMA)
// Conv(8640-tap) as Toeplitz matmul on v_wmma_f32_16x16x32_bf16 with
// split-bf16 (hi+lo) error compensation: 3 WMMAs per tile per K-chunk of 32.
// 2 output tiles per wave to amortize Toeplitz-fragment bandwidth.
// ---------------------------------------------------------------------------

#define T_LEN   262144
#define BATCH   16
#define WINDOW  8640
#define PAD     4320                 // WINDOW/2
#define NCHUNK  271                  // ceil((WINDOW + 15) / 32)
#define SEG     4096                 // outputs per workgroup (8 waves x 2 tiles x 256)
#define NSEG    (T_LEN / SEG)        // 64 segments per batch
#define SPAN    12752                // (SEG-256) + 240 + NCHUNK*32
#define NV4     (SPAN / 4)           // 3188 float4 staging slots
#define SITER   13                   // ceil(NV4 / 256)

typedef __attribute__((ext_vector_type(16))) __bf16       v16bf;
typedef __attribute__((ext_vector_type(4)))  __bf16       v4bf;
typedef __attribute__((ext_vector_type(8)))  float        v8f;
typedef __attribute__((ext_vector_type(4)))  unsigned int u32x4;

union V16 { v16bf v; u32x4 u[2]; };

__device__ __forceinline__ int clampi(int x, int lo, int hi) {
    return x < lo ? lo : (x > hi ? hi : x);
}

// --------------------------- setup: softmax denom ---------------------------
__global__ __launch_bounds__(256)
void k_softmax_denom(const float* __restrict__ tw, float* __restrict__ invZ) {
    __shared__ float red[256];
    float s = 0.f;
    for (int i = threadIdx.x; i < WINDOW; i += 256) s += expf(tw[i]);
    red[threadIdx.x] = s;
    __syncthreads();
    for (int o = 128; o > 0; o >>= 1) {
        if ((int)threadIdx.x < o) red[threadIdx.x] += red[threadIdx.x + o];
        __syncthreads();
    }
    if (threadIdx.x == 0) invZ[0] = 1.0f / red[0];
}

// ---------- setup: Toeplitz B fragments in per-lane WMMA layout -------------
// B[k][n] = c[kbase + k - n], c[j] = softmax(tw)[WINDOW-1-j], zero outside.
// bf16 B 32x16 layout: lanes 0-15 -> N=lane, K=0..7 (elems 0..7), K=16..23 (elems 8..15);
//                      lanes 16-31 -> N=lane-16, K=8..15, K=24..31.
// Flat storage: frag[(chunk*32 + lane)*16 + e], split hi/lo.
__global__ __launch_bounds__(256)
void k_build_frags(const float* __restrict__ tw, const float* __restrict__ invZ,
                   __bf16* __restrict__ fh, __bf16* __restrict__ fl) {
    int gid = blockIdx.x * 256 + threadIdx.x;
    if (gid >= NCHUNK * 512) return;
    int e    = gid & 15;
    int lane = (gid >> 4) & 31;
    int c    = gid >> 9;
    int n    = lane & 15;
    int k    = ((lane >> 4) << 3) + (e & 7) + ((e >> 3) << 4);
    int j    = c * 32 + k - n;
    float v = 0.f;
    if (j >= 0 && j < WINDOW) v = expf(tw[WINDOW - 1 - j]) * invZ[0];
    __bf16 h = (__bf16)v;
    fh[gid] = h;
    fl[gid] = (__bf16)(v - (float)h);
}

// --------------------------- main: Toeplitz WMMA ----------------------------
__global__ __launch_bounds__(256)
void k_baseline_conv_wmma(const float* __restrict__ att,
                          const __bf16* __restrict__ frag_hi,
                          const __bf16* __restrict__ frag_lo,
                          float* __restrict__ out,
                          float* __restrict__ partials) {
    __shared__ __align__(16) __bf16 sh_hi[SPAN];
    __shared__ __align__(16) __bf16 sh_lo[SPAN];
    __shared__ float s_wsum[8];

    const int wg   = blockIdx.x;
    const int b    = wg / NSEG;
    const int seg  = wg % NSEG;
    const int base = seg * SEG;
    const int tid  = threadIdx.x;
    const float* abatch = att + (size_t)b * T_LEN;

    // ---- register-batched staging: issue all loads, then convert ----
    float4 vv[SITER];
    const bool interior = (base - PAD >= 0) && (base - PAD + SPAN <= T_LEN);
    if (interior) {
        const float4* src = (const float4*)(abatch + (base - PAD));
#pragma unroll
        for (int it = 0; it < SITER; ++it) {
            int idx = tid + it * 256;
            if (idx < NV4) vv[it] = src[idx];
        }
    } else {
#pragma unroll
        for (int it = 0; it < SITER; ++it) {
            int idx = tid + it * 256;
            if (idx < NV4) {
                int g0 = base - PAD + idx * 4;
                vv[it].x = abatch[clampi(g0 + 0, 0, T_LEN - 1)];
                vv[it].y = abatch[clampi(g0 + 1, 0, T_LEN - 1)];
                vv[it].z = abatch[clampi(g0 + 2, 0, T_LEN - 1)];
                vv[it].w = abatch[clampi(g0 + 3, 0, T_LEN - 1)];
            }
        }
    }
#pragma unroll
    for (int it = 0; it < SITER; ++it) {
        int idx = tid + it * 256;
        if (idx < NV4) {
            float4 f = vv[it];
            v4bf h, l;
            h.x = (__bf16)f.x; h.y = (__bf16)f.y;
            h.z = (__bf16)f.z; h.w = (__bf16)f.w;
            l.x = (__bf16)(f.x - (float)h.x);
            l.y = (__bf16)(f.y - (float)h.y);
            l.z = (__bf16)(f.z - (float)h.z);
            l.w = (__bf16)(f.w - (float)h.w);
            *(v4bf*)(&sh_hi[idx * 4]) = h;
            *(v4bf*)(&sh_lo[idx * 4]) = l;
        }
    }
    __syncthreads();

    const int wave = tid >> 5;
    const int lane = tid & 31;
    const int tbl  = wave * 512;            // wave's 2-tile offset in segment
    const int M    = lane & 15;             // A-matrix row for this lane
    const int klo  = (lane >> 4) << 3;      // 0 (lanes 0-15) or 8 (lanes 16-31)
    const int off0 = tbl + 16 * M + klo;

    v8f a00 = {}, a01 = {}, a02 = {};
    v8f a10 = {}, a11 = {}, a12 = {};
    const __bf16* fh = frag_hi + (size_t)lane * 16;
    const __bf16* fl = frag_lo + (size_t)lane * 16;

    for (int c = 0; c < NCHUNK; ++c) {
        const int off = off0 + c * 32;
        // A fragments (16-bit A 16x32 layout): elems 0..7 = K klo..klo+7,
        // elems 8..15 = K 16+klo..23+klo. Tile1 is tile0 shifted by 256.
        V16 ah0, al0, ah1, al1;
        ah0.u[0] = *(const u32x4*)(&sh_hi[off]);
        ah0.u[1] = *(const u32x4*)(&sh_hi[off + 16]);
        al0.u[0] = *(const u32x4*)(&sh_lo[off]);
        al0.u[1] = *(const u32x4*)(&sh_lo[off + 16]);
        ah1.u[0] = *(const u32x4*)(&sh_hi[off + 256]);
        ah1.u[1] = *(const u32x4*)(&sh_hi[off + 272]);
        al1.u[0] = *(const u32x4*)(&sh_lo[off + 256]);
        al1.u[1] = *(const u32x4*)(&sh_lo[off + 272]);
        // Shared B fragments: contiguous 32B per lane per chunk.
        v16bf bh = *(const v16bf*)(fh + (size_t)c * 512);
        v16bf bl = *(const v16bf*)(fl + (size_t)c * 512);
        a00 = __builtin_amdgcn_wmma_f32_16x16x32_bf16(false, ah0.v, false, bh,
                                                      (short)0, a00, false, false);
        a10 = __builtin_amdgcn_wmma_f32_16x16x32_bf16(false, ah1.v, false, bh,
                                                      (short)0, a10, false, false);
        a01 = __builtin_amdgcn_wmma_f32_16x16x32_bf16(false, al0.v, false, bh,
                                                      (short)0, a01, false, false);
        a11 = __builtin_amdgcn_wmma_f32_16x16x32_bf16(false, al1.v, false, bh,
                                                      (short)0, a11, false, false);
        a02 = __builtin_amdgcn_wmma_f32_16x16x32_bf16(false, ah0.v, false, bl,
                                                      (short)0, a02, false, false);
        a12 = __builtin_amdgcn_wmma_f32_16x16x32_bf16(false, ah1.v, false, bl,
                                                      (short)0, a12, false, false);
    }
    v8f r0 = a00 + a01 + a02;
    v8f r1 = a10 + a11 + a12;

    // D layout: lanes 0-15 -> N=lane, M=row; lanes 16-31 -> N=lane-16, M=row+8.
    const int n  = lane & 15;
    const int mh = (lane >> 4) * 8;
    float* obase = out + (size_t)b * T_LEN + base + tbl;
    float s = 0.f;
#pragma unroll
    for (int q = 0; q < 8; ++q) {
        obase[16 * (mh + q) + n] = r0[q];
        obase[256 + 16 * (mh + q) + n] = r1[q];
        s += r0[q] + r1[q];
    }
    // Deterministic wave32 tree reduction for the per-wave sum.
#pragma unroll
    for (int o = 16; o > 0; o >>= 1) s += __shfl_xor(s, o, 32);
    if (lane == 0) s_wsum[wave] = s;
    __syncthreads();
    if (tid == 0) {
        float t = 0.f;
#pragma unroll
        for (int q = 0; q < 8; ++q) t += s_wsum[q];
        partials[wg] = t;
    }
}

// ------------------------- per-batch MLP correction -------------------------
__global__ void k_mlp_scale(const float* __restrict__ partials,
                            const float* __restrict__ meta,
                            const float* __restrict__ W1, const float* __restrict__ b1,
                            const float* __restrict__ W2, const float* __restrict__ b2,
                            const float* __restrict__ W3, const float* __restrict__ b3,
                            const float* __restrict__ cs,
                            float* __restrict__ scale) {
    int b = threadIdx.x;
    if (b >= BATCH) return;
    float sum = 0.f;
    for (int i = 0; i < NSEG; ++i) sum += partials[b * NSEG + i];  // fixed order
    float ci[9];
    ci[0] = sum / (float)T_LEN;
    for (int i = 0; i < 8; ++i) ci[1 + i] = meta[b * 8 + i];
    float h1[32];
    for (int o = 0; o < 32; ++o) {
        float a = b1[o];
        for (int i = 0; i < 9; ++i) a += W1[o * 9 + i] * ci[i];
        h1[o] = a > 0.f ? a : 0.f;
    }
    float h2[16];
    for (int o = 0; o < 16; ++o) {
        float a = b2[o];
        for (int i = 0; i < 32; ++i) a += W2[o * 32 + i] * h1[i];
        h2[o] = a > 0.f ? a : 0.f;
    }
    float a = b3[0];
    for (int i = 0; i < 16; ++i) a += W3[i] * h2[i];
    scale[b] = 1.0f + cs[0] * tanhf(a);
}

// ------------------------- apply gain in place ------------------------------
__global__ __launch_bounds__(256)
void k_apply_scale(float* __restrict__ out, const float* __restrict__ scale) {
    int idx = blockIdx.x * 256 + threadIdx.x;       // float4 index
    int b = (idx * 4) / T_LEN;                      // T_LEN % 4 == 0
    float4* p = (float4*)out + idx;
    float4 v = *p;
    float sc = scale[b];
    v.x *= sc; v.y *= sc; v.z *= sc; v.w *= sc;
    *p = v;
}

// ---------------------------------------------------------------------------
extern "C" void kernel_launch(void* const* d_in, const int* in_sizes, int n_in,
                              void* d_out, int out_size, void* d_ws, size_t ws_size,
                              hipStream_t stream) {
    const float* att  = (const float*)d_in[0];
    const float* meta = (const float*)d_in[1];
    const float* tw   = (const float*)d_in[2];
    const float* W1   = (const float*)d_in[3];
    const float* b1   = (const float*)d_in[4];
    const float* W2   = (const float*)d_in[5];
    const float* b2   = (const float*)d_in[6];
    const float* W3   = (const float*)d_in[7];
    const float* b3   = (const float*)d_in[8];
    const float* cs   = (const float*)d_in[9];
    float* out = (float*)d_out;

    char* ws = (char*)d_ws;
    __bf16* frag_hi = (__bf16*)(ws + 0);        // NCHUNK*512*2 = 277504 B
    __bf16* frag_lo = (__bf16*)(ws + 294912);   // 277504 B
    float*  invZ    = (float*)(ws + 589824);    // 4 B
    float*  partials= (float*)(ws + 590080);    // BATCH*NSEG*4 = 4096 B
    float*  scale   = (float*)(ws + 598528);    // 64 B

    k_softmax_denom<<<1, 256, 0, stream>>>(tw, invZ);
    k_build_frags<<<(NCHUNK * 512 + 255) / 256, 256, 0, stream>>>(tw, invZ,
                                                                  frag_hi, frag_lo);
    k_baseline_conv_wmma<<<BATCH * NSEG, 256, 0, stream>>>(att, frag_hi, frag_lo,
                                                           out, partials);
    k_mlp_scale<<<1, 32, 0, stream>>>(partials, meta, W1, b1, W2, b2, W3, b3,
                                      cs, scale);
    k_apply_scale<<<(BATCH * T_LEN / 4 + 255) / 256, 256, 0, stream>>>(out, scale);
}
